// MultiHeadSelfAttentionBlock_56556129353873
// MI455X (gfx1250) — matrix operand order repaired and checked by reading, not compiled, not run on hardware
//
#include <hip/hip_runtime.h>
#include <hip/hip_bf16.h>

// ---------------------------------------------------------------------------
// MultiHeadSelfAttentionBlock for MI455X (gfx1250, wave32, WMMA)
//   N=50000 pts, C=128, H=8 heads, D=16, KV=27, M=800000 edges
//   Dense GEMMs ([N,128]x[128,128]) via v_wmma_f32_16x16x32_f16 with
//   pre-swizzled operand layouts (fragments are contiguous 32B loads).
// ---------------------------------------------------------------------------

typedef __attribute__((ext_vector_type(16))) _Float16 v16h;
typedef __attribute__((ext_vector_type(8)))  _Float16 v8h;
typedef __attribute__((ext_vector_type(8)))  float    v8f;

#define CCH 128          // channels
#define HEADS 8
#define HD 16            // head dim
#define BN_EPS 1e-5f

// ---------------------------------------------------------------------------
// WMMA fragment swizzle index helpers (CDNA5 ISA 7.12.2, 16-bit A/B layouts)
//   A 16x32 f16: lane = row (lane&15); lanes<16 hold K {0..7,16..23},
//                lanes>=16 hold K {8..15,24..31}; a[i] -> K = hi*8+(i&7)+((i>>3)<<4)
//   B 32x16 f16: lane = col (lane&15); same K split as A;
//                b[i] -> K = hi*8+(i&7)+((i>>3)<<4)
// Swizzled storage: frag[(tile*32 + lane)*16 + i] contiguous per lane.
// ---------------------------------------------------------------------------
__device__ __forceinline__ int swzA_idx(int row, int k) {
  int rowTile = row >> 4, lr = row & 15;
  int kt = k >> 5, kk = k & 31;
  int i = ((kk >> 4) << 3) | (kk & 7);
  int hi = (kk >> 3) & 1;
  int lane = (hi << 4) | lr;
  return (((rowTile << 2) + kt) << 9) + (lane << 4) + i;
}
__device__ __forceinline__ int swzB_idx(int k, int col) {
  int w = col >> 4, lc = col & 15;
  int kt = k >> 5, kk = k & 31;
  int hi = (kk >> 3) & 1, i = ((kk >> 4) << 3) | (kk & 7);
  int lane = (hi << 4) | lc;
  return (((w << 2) + kt) << 9) + (lane << 4) + i;
}

// f32 [nrows x 128] -> swizzled-A f16
__global__ void cvt_swzA_kernel(const float* __restrict__ src,
                                _Float16* __restrict__ dst, int n) {
  int idx = blockIdx.x * blockDim.x + threadIdx.x;
  if (idx >= n) return;
  int row = idx >> 7, k = idx & (CCH - 1);
  dst[swzA_idx(row, k)] = (_Float16)src[idx];
}

// f32 [128 x 128] weight -> swizzled-B f16
__global__ void cvt_swzB_kernel(const float* __restrict__ src,
                                _Float16* __restrict__ dst) {
  int idx = blockIdx.x * blockDim.x + threadIdx.x;
  if (idx >= CCH * CCH) return;
  int k = idx >> 7, col = idx & (CCH - 1);
  dst[swzB_idx(k, col)] = (_Float16)src[idx];
}

// ---------------------------------------------------------------------------
// WMMA GEMM:  Cout[N x 128] = A[N x 128] @ B[128 x 128] (+ bias)
// Block = 8 waves; wave w owns column tile w, computes TWO 16x16 row tiles
// sharing each B fragment (branch-free inner loop: the odd-tail block clamps
// tile1 to tile0 and only guards the store). B (32 KB swizzled f16) staged
// in LDS; every fragment is one contiguous 32B load.
// C/D layout: VGPR j -> row = (lane>>4)*8 + j, col = lane&15.
// ---------------------------------------------------------------------------
__global__ __launch_bounds__(256)
void gemm_wmma_kernel(const _Float16* __restrict__ Aswz,
                      const _Float16* __restrict__ Bswz,
                      const float* __restrict__ bias,
                      float* __restrict__ Cout, int nrows) {
  __shared__ _Float16 Bs[CCH * CCH];
  {
    const v8h* src = (const v8h*)Bswz;
    v8h* dst = (v8h*)Bs;
    #pragma unroll 2
    for (int i = threadIdx.x; i < (CCH * CCH) / 8; i += 256) dst[i] = src[i];
  }
  __syncthreads();

  const int wv   = threadIdx.x >> 5;   // wave id = column tile (0..7)
  const int lane = threadIdx.x & 31;
  const int half = lane >> 4;
  const int l    = lane & 15;

  const int numRT = nrows >> 4;
  const int t0 = blockIdx.x * 2;
  const bool has1 = (t0 + 1 < numRT);  // wave-uniform; only final block is odd
  const int t1 = has1 ? (t0 + 1) : t0; // clamp -> branch-free inner loop

  const _Float16* a0p = Aswz + ((size_t)t0 << 11) + (lane << 4); // t0*4*512
  const _Float16* a1p = Aswz + ((size_t)t1 << 11) + (lane << 4);
  __builtin_prefetch(a0p, 0, 3);       // global_prefetch_b8

  v8f acc0 = {}, acc1 = {};
  #pragma unroll
  for (int kt = 0; kt < 4; ++kt) {
    v16h b  = *(const v16h*)(&Bs[(((wv << 2) + kt) << 9) + (lane << 4)]);
    v16h a0 = *(const v16h*)(a0p + (kt << 9));
    v16h a1 = *(const v16h*)(a1p + (kt << 9));
    acc0 = __builtin_amdgcn_wmma_f32_16x16x32_f16(
        false, a0, false, b, (short)0, acc0, false, false);
    acc1 = __builtin_amdgcn_wmma_f32_16x16x32_f16(
        false, a1, false, b, (short)0, acc1, false, false);
  }

  const int ncol = (wv << 4) + l;
  const float bvv = bias ? bias[ncol] : 0.0f;
  {
    float* cp = Cout + (size_t)(t0 * 16 + half * 8) * CCH + ncol;
    #pragma unroll
    for (int j = 0; j < 8; ++j) cp[(size_t)j * CCH] = acc0[j] + bvv;
  }
  if (has1) {
    float* cp = Cout + (size_t)(t1 * 16 + half * 8) * CCH + ncol;
    #pragma unroll
    for (int j = 0; j < 8; ++j) cp[(size_t)j * CCH] = acc1[j] + bvv;
  }
}

// ---------------------------------------------------------------------------
// Relative-position encoder: r[27,128] = relu(rp@pW1+pb1)@pW2+pb2
// ---------------------------------------------------------------------------
__global__ void relpos_kernel(const float* __restrict__ pW1,
                              const float* __restrict__ pb1,
                              const float* __restrict__ pW2,
                              const float* __restrict__ pb2,
                              float* __restrict__ r) {
  int i = blockIdx.x * blockDim.x + threadIdx.x;
  if (i >= 27 * CCH) return;
  int kv = i >> 7, c = i & (CCH - 1);
  float in[3];
  in[0] = 1.0f - (float)(kv % 3);        // px
  in[1] = 1.0f - (float)((kv / 3) % 3);  // py
  in[2] = 1.0f - (float)(kv / 9);        // pz
  float acc = pb2[c];
  #pragma unroll
  for (int j = 0; j < 3; ++j) {
    float t = pb1[j];
    #pragma unroll
    for (int ii = 0; ii < 3; ++ii) t += in[ii] * pW1[ii * 3 + j];
    t = fmaxf(t, 0.0f);
    acc += t * pW2[j * CCH + c];
  }
  r[i] = acc;
}

// order-preserving float<->uint encoding for atomicMax over signed floats
__device__ __forceinline__ unsigned enc_f32(float f) {
  unsigned u = __float_as_uint(f);
  return (u & 0x80000000u) ? ~u : (u | 0x80000000u);
}
__device__ __forceinline__ float dec_f32(unsigned u) {
  return __uint_as_float((u & 0x80000000u) ? (u ^ 0x80000000u) : ~u);
}

// logits[m,h] = <q[query,h,:], k[key,h,:]+r[kern,h,:]> / 4  (+ seg max)
__global__ void logits_kernel(const float* __restrict__ q,
                              const float* __restrict__ k,
                              const float* __restrict__ r,
                              const int* __restrict__ query_idx,
                              const int* __restrict__ key_idx,
                              const int* __restrict__ kernel_idx,
                              float* __restrict__ logits,
                              unsigned* __restrict__ segmax, int M) {
  int idx = blockIdx.x * blockDim.x + threadIdx.x;
  if (idx >= M * HEADS) return;
  int m = idx >> 3, h = idx & (HEADS - 1);
  int qi = query_idx[m], ki = key_idx[m], kk = kernel_idx[m];
  const float4* qp = (const float4*)(q + (size_t)qi * CCH + h * HD);
  const float4* kp = (const float4*)(k + (size_t)ki * CCH + h * HD);
  const float4* rp = (const float4*)(r + (size_t)kk * CCH + h * HD);
  float s = 0.0f;
  #pragma unroll
  for (int t = 0; t < 4; ++t) {
    float4 a = qp[t], b = kp[t], c = rp[t];
    s += a.x * (b.x + c.x) + a.y * (b.y + c.y) +
         a.z * (b.z + c.z) + a.w * (b.w + c.w);
  }
  s *= 0.25f;  // 1/sqrt(D), D=16
  logits[idx] = s;
  atomicMax(&segmax[qi * HEADS + h], enc_f32(s));
}

// e = exp(logit - segmax); denom += e
__global__ void softmax_exp_kernel(float* __restrict__ logits,
                                   const unsigned* __restrict__ segmax,
                                   float* __restrict__ denom,
                                   const int* __restrict__ query_idx, int M) {
  int idx = blockIdx.x * blockDim.x + threadIdx.x;
  if (idx >= M * HEADS) return;
  int m = idx >> 3, h = idx & (HEADS - 1);
  int qi = query_idx[m];
  float mx = dec_f32(segmax[qi * HEADS + h]);
  float e = __expf(logits[idx] - mx);
  logits[idx] = e;
  atomicAdd(&denom[qi * HEADS + h], e);
}

// out[query,h,:] += (e/denom) * v[key,h,:]
__global__ void scatter_av_kernel(const float* __restrict__ e,
                                  const float* __restrict__ denom,
                                  const float* __restrict__ v,
                                  const int* __restrict__ query_idx,
                                  const int* __restrict__ key_idx,
                                  float* __restrict__ outacc, int M) {
  int idx = blockIdx.x * blockDim.x + threadIdx.x;
  if (idx >= M * HEADS) return;
  int m = idx >> 3, h = idx & (HEADS - 1);
  int qi = query_idx[m], ki = key_idx[m];
  float attn = e[idx] / denom[qi * HEADS + h];
  const float4* vp = (const float4*)(v + (size_t)ki * CCH + h * HD);
  float* op = outacc + (size_t)qi * CCH + h * HD;
  #pragma unroll
  for (int t = 0; t < 4; ++t) {
    float4 vv = vp[t];
    atomicAdd(op + 4 * t + 0, attn * vv.x);
    atomicAdd(op + 4 * t + 1, attn * vv.y);
    atomicAdd(op + 4 * t + 2, attn * vv.z);
    atomicAdd(op + 4 * t + 3, attn * vv.w);
  }
}

// ---------------------------------------------------------------------------
// BatchNorm (training stats, biased variance like jnp.var)
// ---------------------------------------------------------------------------
__global__ void bn_stats_kernel(const float* __restrict__ a,
                                float* __restrict__ sums,
                                float* __restrict__ sumsq, int nrows) {
  int c = threadIdx.x;  // blockDim.x == 128
  int r0 = blockIdx.x * 256;
  int r1 = min(r0 + 256, nrows);
  float s = 0.0f, s2 = 0.0f;
  for (int r = r0; r < r1; ++r) {
    float v = a[(size_t)r * CCH + c];
    s += v; s2 += v * v;
  }
  atomicAdd(&sums[c], s);
  atomicAdd(&sumsq[c], s2);
}

// y = BN(a)*gamma+beta [; relu] [; +resid] -> optional f32 out,
// optional swizzled-A f16 out (feeds the next WMMA GEMM directly)
__global__ void bn_apply_kernel(const float* __restrict__ a,
                                const float* __restrict__ sums,
                                const float* __restrict__ sumsq,
                                const float* __restrict__ gamma,
                                const float* __restrict__ beta,
                                const float* __restrict__ resid,
                                float* __restrict__ outf,
                                _Float16* __restrict__ outh_swz,
                                int nrows, int do_relu) {
  int i = blockIdx.x * blockDim.x + threadIdx.x;
  if (i >= nrows * CCH) return;
  int c = i & (CCH - 1);
  float invn = 1.0f / (float)nrows;
  float mean = sums[c] * invn;
  float var = fmaxf(sumsq[c] * invn - mean * mean, 0.0f);
  float rstd = rsqrtf(var + BN_EPS);
  float val = (a[i] - mean) * rstd * gamma[c] + beta[c];
  if (do_relu) val = fmaxf(val, 0.0f);
  if (resid) val += resid[i];
  if (outf) outf[i] = val;
  if (outh_swz) outh_swz[swzA_idx(i >> 7, c)] = (_Float16)val;
}

// ---------------------------------------------------------------------------
extern "C" void kernel_launch(void* const* d_in, const int* in_sizes, int n_in,
                              void* d_out, int out_size, void* d_ws, size_t ws_size,
                              hipStream_t stream) {
  const float* x          = (const float*)d_in[0];
  const int*   key_idx    = (const int*)d_in[1];
  const int*   kernel_idx = (const int*)d_in[2];
  const int*   query_idx  = (const int*)d_in[3];
  const float* Wq = (const float*)d_in[4];  const float* bq = (const float*)d_in[5];
  const float* Wk = (const float*)d_in[6];  const float* bk = (const float*)d_in[7];
  const float* Wv = (const float*)d_in[8];  const float* bv = (const float*)d_in[9];
  const float* Wo = (const float*)d_in[10]; const float* bo = (const float*)d_in[11];
  const float* pW1 = (const float*)d_in[12]; const float* pb1 = (const float*)d_in[13];
  const float* pW2 = (const float*)d_in[14]; const float* pb2 = (const float*)d_in[15];
  const float* g1  = (const float*)d_in[16]; const float* be1 = (const float*)d_in[17];
  const float* mW1 = (const float*)d_in[18];
  const float* mg  = (const float*)d_in[19]; const float* mb  = (const float*)d_in[20];
  const float* mW2 = (const float*)d_in[21]; const float* mb2 = (const float*)d_in[22];
  const float* g2  = (const float*)d_in[23]; const float* be2 = (const float*)d_in[24];

  const int Nn = in_sizes[0] / CCH;     // 50000 (divisible by 16)
  const int M  = in_sizes[1];           // 800000
  const size_t NC = (size_t)Nn * CCH;
  const int MH = M * HEADS;

  // ---- workspace carve-up (256B aligned) ----
  char* p = (char*)d_ws;
  auto take = [&](size_t bytes) -> void* {
    void* r = (void*)p;
    p += (bytes + 255) & ~(size_t)255;
    return r;
  };
  float*     bufA   = (float*)take(NC * 4);                       // q -> attn accum
  float*     bufB   = (float*)take(NC * 4);                       // k -> Wo output
  float*     bufC   = (float*)take(NC * 4);                       // v -> mlp hidden
  float*     bufF   = (float*)take(NC * 4);                       // out_ (resid)
  size_t dsz = ((size_t)MH > NC ? (size_t)MH : NC);
  float*     bufD   = (float*)take(dsz * 4);                      // logits -> mlp out
  _Float16*  bufE   = (_Float16*)take(NC * 2);                    // swizzled f16 A
  _Float16*  whq    = (_Float16*)take(CCH * CCH * 2);
  _Float16*  whk    = (_Float16*)take(CCH * CCH * 2);
  _Float16*  whv    = (_Float16*)take(CCH * CCH * 2);
  _Float16*  who    = (_Float16*)take(CCH * CCH * 2);
  _Float16*  whm1   = (_Float16*)take(CCH * CCH * 2);
  _Float16*  whm2   = (_Float16*)take(CCH * CCH * 2);
  float*     rbuf   = (float*)take(27 * CCH * 4);
  unsigned*  segmax = (unsigned*)take((size_t)Nn * HEADS * 4);
  float*     denom  = (float*)take((size_t)Nn * HEADS * 4);
  float*     bnsum  = (float*)take(CCH * 4);
  float*     bnss   = (float*)take(CCH * 4);
  (void)ws_size; (void)n_in;

  const int nct   = (int)((NC + 255) / 256);
  const int mht   = (MH + 255) / 256;
  const int numRT = Nn / 16;                 // 3125 row tiles
  const int gblk  = (numRT + 1) / 2;         // 2 row tiles per block
  const int bnblk = (Nn + 255) / 256;
  const int wblk  = (CCH * CCH + 255) / 256;

  // ---- swizzled f16 conversions ----
  cvt_swzA_kernel<<<nct, 256, 0, stream>>>(x, bufE, (int)NC);
  cvt_swzB_kernel<<<wblk, 256, 0, stream>>>(Wq,  whq);
  cvt_swzB_kernel<<<wblk, 256, 0, stream>>>(Wk,  whk);
  cvt_swzB_kernel<<<wblk, 256, 0, stream>>>(Wv,  whv);
  cvt_swzB_kernel<<<wblk, 256, 0, stream>>>(Wo,  who);
  cvt_swzB_kernel<<<wblk, 256, 0, stream>>>(mW1, whm1);
  cvt_swzB_kernel<<<wblk, 256, 0, stream>>>(mW2, whm2);

  // ---- rel-pos table ----
  relpos_kernel<<<(27 * CCH + 255) / 256, 256, 0, stream>>>(pW1, pb1, pW2, pb2, rbuf);

  // ---- Q/K/V projections (WMMA) ----
  gemm_wmma_kernel<<<gblk, 256, 0, stream>>>(bufE, whq, bq, bufA, Nn);
  gemm_wmma_kernel<<<gblk, 256, 0, stream>>>(bufE, whk, bk, bufB, Nn);
  gemm_wmma_kernel<<<gblk, 256, 0, stream>>>(bufE, whv, bv, bufC, Nn);

  // ---- segment softmax attention ----
  hipMemsetAsync(segmax, 0, (size_t)Nn * HEADS * 4, stream);  // 0 < enc(-inf): below any float
  hipMemsetAsync(denom,  0, (size_t)Nn * HEADS * 4, stream);
  logits_kernel<<<mht, 256, 0, stream>>>(bufA, bufB, rbuf, query_idx, key_idx,
                                         kernel_idx, bufD, segmax, M);
  softmax_exp_kernel<<<mht, 256, 0, stream>>>(bufD, segmax, denom, query_idx, M);
  hipMemsetAsync(bufA, 0, NC * 4, stream);                    // reuse q as accum
  scatter_av_kernel<<<mht, 256, 0, stream>>>(bufD, denom, bufC, query_idx,
                                             key_idx, bufA, M);

  // ---- output projection ----
  cvt_swzA_kernel<<<nct, 256, 0, stream>>>(bufA, bufE, (int)NC);
  gemm_wmma_kernel<<<gblk, 256, 0, stream>>>(bufE, who, bo, bufB, Nn);

  // ---- norm1 + residual -> out_ (f32 in bufF, swizzled f16 in bufE) ----
  hipMemsetAsync(bnsum, 0, CCH * 4, stream);
  hipMemsetAsync(bnss,  0, CCH * 4, stream);
  bn_stats_kernel<<<bnblk, 128, 0, stream>>>(bufB, bnsum, bnss, Nn);
  bn_apply_kernel<<<nct, 256, 0, stream>>>(bufB, bnsum, bnss, g1, be1, x,
                                           bufF, bufE, Nn, 0);

  // ---- MLP: Linear(no bias) -> BN -> ReLU -> Linear ----
  gemm_wmma_kernel<<<gblk, 256, 0, stream>>>(bufE, whm1, nullptr, bufC, Nn);
  hipMemsetAsync(bnsum, 0, CCH * 4, stream);
  hipMemsetAsync(bnss,  0, CCH * 4, stream);
  bn_stats_kernel<<<bnblk, 128, 0, stream>>>(bufC, bnsum, bnss, Nn);
  bn_apply_kernel<<<nct, 256, 0, stream>>>(bufC, bnsum, bnss, mg, mb, nullptr,
                                           nullptr, bufE, Nn, 1);
  gemm_wmma_kernel<<<gblk, 256, 0, stream>>>(bufE, whm2, mb2, bufD, Nn);

  // ---- norm2 + residual -> d_out ----
  hipMemsetAsync(bnsum, 0, CCH * 4, stream);
  hipMemsetAsync(bnss,  0, CCH * 4, stream);
  bn_stats_kernel<<<bnblk, 128, 0, stream>>>(bufD, bnsum, bnss, Nn);
  bn_apply_kernel<<<nct, 256, 0, stream>>>(bufD, bnsum, bnss, g2, be2, bufF,
                                           (float*)d_out, nullptr, Nn, 0);
  (void)out_size;
}
